// Decoder_16569983828132
// MI455X (gfx1250) — matrix operand order repaired and checked by reading, not compiled, not run on hardware
//
#include <hip/hip_runtime.h>
#include <hip/hip_bf16.h>
#include <math.h>

// ---------------------------------------------------------------------------
// 6-layer transformer decoder, MI455X (gfx1250, wave32).
//  - all matmuls via v_wmma_f32_16x16x32_f16 (f16 in, f32 accumulate)
//  - weights pre-transposed+converted to f16 once per launch
//  - activations carried as f16 mirrors -> GEMM/attention LDS fills are pure
//    16B async global->LDS copies (ASYNCcnt) with double buffering
//  - flash-attention online softmax (no score tensor materialization)
// ---------------------------------------------------------------------------

typedef _Float16 h8  __attribute__((ext_vector_type(8)));
typedef _Float16 h16 __attribute__((ext_vector_type(16)));
typedef float    f8  __attribute__((ext_vector_type(8)));
typedef int      v4i __attribute__((ext_vector_type(4)));

#define C_B   4
#define C_T   1024
#define C_D   1024
#define C_H   4096
#define C_M   4096            // B*T rows
#define C_NH  16
#define C_DH  64
#define C_INF 1e10f
#define C_INVSCALE (1.0f/32.0f)   // 1/sqrt(1024)

// -------------------- async global->LDS copy (CDNA5) -----------------------
#if defined(__has_builtin)
#  if __has_builtin(__builtin_amdgcn_global_load_async_to_lds_b128)
#    define HAVE_ASYNC_LDS 1
#  endif
#endif

#if defined(HAVE_ASYNC_LDS)
__device__ __forceinline__ void cp16_async(const _Float16* g, _Float16* l) {
  __builtin_amdgcn_global_load_async_to_lds_b128(
      (__attribute__((address_space(1))) v4i*)(g),
      (__attribute__((address_space(3))) v4i*)(l), 0, 0);
}
__device__ __forceinline__ void cp_wait() {
#  if __has_builtin(__builtin_amdgcn_s_wait_asynccnt)
  __builtin_amdgcn_s_wait_asynccnt(0);
#  else
  asm volatile("s_wait_asynccnt 0x0" ::: "memory");
#  endif
}
#else
__device__ __forceinline__ void cp16_async(const _Float16* g, _Float16* l) {
  *(h8*)l = *(const h8*)g;
}
__device__ __forceinline__ void cp_wait() {}
#endif

// -------------------- WMMA fragment helpers --------------------------------
// A fragment (16-bit A 16x32): per lane 16 contiguous K halfs
__device__ __forceinline__ h16 ld_frag_gap(const _Float16* p) {
  h8 lo = *(const h8*)p;
  h8 hi = *(const h8*)(p + 8);
  return __builtin_shufflevector(lo, hi, 0,1,2,3,4,5,6,7,8,9,10,11,12,13,14,15);
}
// B fragment (16-bit B 32x16): per lane 16 contiguous K halfs
__device__ __forceinline__ h16 ld_frag_ct(const _Float16* p) {
  h8 lo = *(const h8*)p;
  h8 hi = *(const h8*)(p + 8);
  return __builtin_shufflevector(lo, hi, 0,1,2,3,4,5,6,7,8,9,10,11,12,13,14,15);
}
__device__ __forceinline__ f8 wmma_f16(h16 a, h16 b, f8 c) {
  return __builtin_amdgcn_wmma_f32_16x16x32_f16(false, a, false, b, (short)0, c, false, false);
}

// ---------------------------------------------------------------------------
// Prep kernels
// ---------------------------------------------------------------------------
// x -> f32 copy + f16 mirror (4 elems/thread)
__global__ __launch_bounds__(256) void init_x_kernel(
    const float* __restrict__ x, float* __restrict__ xa, _Float16* __restrict__ xh)
{
  size_t i = ((size_t)blockIdx.x * 256 + threadIdx.x) * 4;
  float4 f = *(const float4*)(x + i);
  *(float4*)(xa + i) = f;
  _Float16* p = xh + i;
  p[0] = (_Float16)f.x; p[1] = (_Float16)f.y; p[2] = (_Float16)f.z; p[3] = (_Float16)f.w;
}

// flat f32 -> f16 (4 elems/thread)
__global__ __launch_bounds__(256) void cvt_f16_kernel(
    const float* __restrict__ s, _Float16* __restrict__ d)
{
  size_t i = ((size_t)blockIdx.x * 256 + threadIdx.x) * 4;
  float4 f = *(const float4*)(s + i);
  _Float16* p = d + i;
  p[0] = (_Float16)f.x; p[1] = (_Float16)f.y; p[2] = (_Float16)f.z; p[3] = (_Float16)f.w;
}

// W [K x Nc] f32 -> Wt [Nc x K] f16, batched over layers via blockIdx.z
__global__ __launch_bounds__(256) void transpose_cvt_kernel(
    const float* __restrict__ W, _Float16* __restrict__ Wt, int K, int Nc)
{
  __shared__ float t[32][33];
  const size_t slab = (size_t)K * Nc;
  const float*  Wl  = W  + (size_t)blockIdx.z * slab;
  _Float16*     Wtl = Wt + (size_t)blockIdx.z * slab;
  const int k0 = blockIdx.y * 32, n0 = blockIdx.x * 32;
  const int tx = threadIdx.x & 31, ty = threadIdx.x >> 5;  // ty in [0,8)
  #pragma unroll
  for (int i = 0; i < 4; ++i) {
    int k = ty + i * 8;
    t[k][tx] = Wl[(size_t)(k0 + k) * Nc + n0 + tx];
  }
  __syncthreads();
  #pragma unroll
  for (int i = 0; i < 4; ++i) {
    int n = ty + i * 8;
    Wtl[(size_t)(n0 + n) * K + k0 + tx] = (_Float16)t[tx][n];
  }
}

// ---------------------------------------------------------------------------
// GEMM: C[M x Nc] = act(A[M x K] @ W[K x Nc] + bias)
//   A   : f16 [M][K] row-major
//   Wt  : f16 [Nc][K] (pre-transposed) -> B tile lands in LDS as [n][k]
// Tile 128x128x64, 256 threads = 8 waves (4 x 2), wave = 32x64 (2x4 WMMA),
// double-buffered LDS filled with async b128 copies.
// ---------------------------------------------------------------------------
#define GM_BM 128
#define GM_BN 128
#define GM_BK 64
#define GM_LD 72      // halves stride (64 + 8 pad)

__device__ __forceinline__ void gemm_prefetch(
    const _Float16* __restrict__ A, const _Float16* __restrict__ Wt,
    _Float16* As, _Float16* Bs, int bm, int bn, int K, int k0, int tid)
{
  #pragma unroll
  for (int it = 0; it < 4; ++it) {                  // A: 128 rows x 64 halfs
    int i = tid + it * 256, r = i >> 3, c8 = (i & 7) << 3;
    cp16_async(A + (size_t)(bm + r) * K + k0 + c8, &As[r * GM_LD + c8]);
  }
  #pragma unroll
  for (int it = 0; it < 4; ++it) {                  // B: 128 n-rows x 64 halfs
    int i = tid + it * 256, r = i >> 3, c8 = (i & 7) << 3;
    cp16_async(Wt + (size_t)(bn + r) * K + k0 + c8, &Bs[r * GM_LD + c8]);
  }
}

template<int ACT, int OUTF16>
__global__ __launch_bounds__(256) void gemm_kernel(
    const _Float16* __restrict__ A, const _Float16* __restrict__ Wt,
    const float* __restrict__ bias,
    float* __restrict__ Cf, _Float16* __restrict__ Ch,
    int M, int K, int Nc)
{
  __shared__ __attribute__((aligned(16))) _Float16 Asl[2][GM_BM * GM_LD];
  __shared__ __attribute__((aligned(16))) _Float16 Bsl[2][GM_BN * GM_LD];

  const int tid  = threadIdx.x;
  const int lane = tid & 31;
  const int wave = tid >> 5;
  const int wm   = wave & 3;       // 32-row band
  const int wn   = wave >> 2;      // 64-col band
  const int l15  = lane & 15;
  const int hsel = (lane < 16) ? 0 : 1;
  const int bm   = blockIdx.y * GM_BM;
  const int bn   = blockIdx.x * GM_BN;

  f8 acc[2][4] = {};

  const int nK = K / GM_BK;
  gemm_prefetch(A, Wt, Asl[0], Bsl[0], bm, bn, K, 0, tid);
  cp_wait();
  __syncthreads();

  for (int kt = 0; kt < nK; ++kt) {
    const int cur = kt & 1;
    if (kt + 1 < nK)
      gemm_prefetch(A, Wt, Asl[cur ^ 1], Bsl[cur ^ 1], bm, bn, K, (kt + 1) * GM_BK, tid);

    #pragma unroll
    for (int ks = 0; ks < 2; ++ks) {
      h16 af[2], bf[4];
      #pragma unroll
      for (int mi = 0; mi < 2; ++mi)
        af[mi] = ld_frag_gap(&Asl[cur][(wm * 32 + mi * 16 + l15) * GM_LD + ks * 32 + hsel * 16]);
      #pragma unroll
      for (int ni = 0; ni < 4; ++ni)
        bf[ni] = ld_frag_ct(&Bsl[cur][(wn * 64 + ni * 16 + l15) * GM_LD + ks * 32 + hsel * 16]);
      #pragma unroll
      for (int mi = 0; mi < 2; ++mi)
        #pragma unroll
        for (int ni = 0; ni < 4; ++ni)
          acc[mi][ni] = wmma_f16(af[mi], bf[ni], acc[mi][ni]);
    }
    cp_wait();        // next tile resident (this wave's copies)
    __syncthreads();  // all waves done with cur + all copies visible
  }

  // epilogue: C layout: lane<16 -> rows r, lane>=16 -> rows 8+r; col = lane&15
  #pragma unroll
  for (int mi = 0; mi < 2; ++mi) {
    #pragma unroll
    for (int ni = 0; ni < 4; ++ni) {
      int col  = bn + wn * 64 + ni * 16 + l15;
      float bv = bias[col];
      #pragma unroll
      for (int r = 0; r < 8; ++r) {
        int row = bm + wm * 32 + mi * 16 + (hsel ? 8 + r : r);
        float v = acc[mi][ni][r] + bv;
        if (ACT == 1) v = v > 0.0f ? v : 0.0f;
        if (OUTF16) Ch[(size_t)row * Nc + col] = (_Float16)v;
        else        Cf[(size_t)row * Nc + col] = v;
      }
    }
  }
}

// ---------------------------------------------------------------------------
// Flash attention over f16 Q/K/V, f16 output. One WG per (b*16+h, 64-row
// q-tile); 128 threads = 4 waves; wave w owns q rows [16w,16w+16).
// ---------------------------------------------------------------------------
#define AT_LD 72   // halves (64 + 8 pad)

__global__ __launch_bounds__(128) void attn_kernel(
    const _Float16* __restrict__ Q, const _Float16* __restrict__ Kb,
    const _Float16* __restrict__ Vb, const float* __restrict__ mask,
    _Float16* __restrict__ O, int causal)
{
  __shared__ __attribute__((aligned(16))) _Float16 Qs[64 * AT_LD];
  __shared__ __attribute__((aligned(16))) _Float16 Ks[64 * AT_LD];
  __shared__ __attribute__((aligned(16))) _Float16 Vs[64 * AT_LD];  // [d][key]
  __shared__ __attribute__((aligned(16))) _Float16 Ps[64 * AT_LD];

  const int tid  = threadIdx.x;
  const int lane = tid & 31;
  const int wq   = tid >> 5;
  const int l15  = lane & 15;
  const int hsel = (lane < 16) ? 0 : 1;

  const int qt = blockIdx.x;
  const int bh = blockIdx.y;
  const int b  = bh >> 4;
  const int h  = bh & 15;
  const int hc = h * C_DH;
  const int qrow0 = b * C_T + qt * 64;
  const int krow0 = b * C_T;
  const float* maskp = mask + (size_t)b * C_T;

  // async-load Q tile (64 rows x 64 halfs = 512 x 16B chunks)
  #pragma unroll
  for (int it = 0; it < 4; ++it) {
    int i = tid + it * 128, r = i >> 3, c8 = (i & 7) << 3;
    cp16_async(Q + (size_t)(qrow0 + r) * C_D + hc + c8, &Qs[r * AT_LD + c8]);
  }

  float m[8], l[8];
  f8 oacc[4] = {};
  #pragma unroll
  for (int r = 0; r < 8; ++r) { m[r] = -1e30f; l[r] = 0.0f; }

  const int nkt = causal ? (qt + 1) : (C_T / 64);
  for (int kt = 0; kt < nkt; ++kt) {
    __syncthreads();   // previous iteration done with Ks/Vs
    // K tile async [key][d]
    #pragma unroll
    for (int it = 0; it < 4; ++it) {
      int i = tid + it * 128, r = i >> 3, c8 = (i & 7) << 3;
      cp16_async(Kb + (size_t)(krow0 + kt * 64 + r) * C_D + hc + c8, &Ks[r * AT_LD + c8]);
    }
    // V tile manual transpose -> [d][key]
    #pragma unroll
    for (int it = 0; it < 4; ++it) {
      int i = tid + it * 128, r = i >> 3, c8 = (i & 7) << 3;
      h8 v = *(const h8*)(Vb + (size_t)(krow0 + kt * 64 + r) * C_D + hc + c8);
      #pragma unroll
      for (int j = 0; j < 8; ++j) Vs[(c8 + j) * AT_LD + r] = v[j];
    }
    cp_wait();
    __syncthreads();

    // ---- S = Q @ K^T ----
    f8 sacc[4] = {};
    #pragma unroll
    for (int kd = 0; kd < 2; ++kd) {
      h16 qa = ld_frag_gap(&Qs[(wq * 16 + l15) * AT_LD + kd * 32 + hsel * 16]);
      #pragma unroll
      for (int nt = 0; nt < 4; ++nt) {
        h16 kf = ld_frag_ct(&Ks[(nt * 16 + l15) * AT_LD + kd * 32 + hsel * 16]);
        sacc[nt] = wmma_f16(qa, kf, sacc[nt]);
      }
    }

    // ---- mask + online softmax ----
    float kmask[4];
    int   kcol[4];
    #pragma unroll
    for (int nt = 0; nt < 4; ++nt) {
      kcol[nt]  = kt * 64 + nt * 16 + l15;
      kmask[nt] = maskp[kcol[nt]];
    }
    float p[4][8], mloc[8];
    #pragma unroll
    for (int r = 0; r < 8; ++r) {
      int qrow = qt * 64 + wq * 16 + (hsel ? 8 + r : r);
      float mx = -1e30f;
      #pragma unroll
      for (int nt = 0; nt < 4; ++nt) {
        float sv = sacc[nt][r];
        sv -= (1.0f - kmask[nt]) * C_INF;
        if (causal && kcol[nt] > qrow) sv -= C_INF;
        sv *= C_INVSCALE;
        p[nt][r] = sv;
        mx = fmaxf(mx, sv);
      }
      mloc[r] = mx;
    }
    #pragma unroll
    for (int r = 0; r < 8; ++r)
      #pragma unroll
      for (int off = 1; off < 16; off <<= 1)
        mloc[r] = fmaxf(mloc[r], __shfl_xor(mloc[r], off));

    float corr[8], rs[8];
    #pragma unroll
    for (int r = 0; r < 8; ++r) {
      float mnew = fmaxf(m[r], mloc[r]);
      corr[r] = expf(m[r] - mnew);
      m[r] = mnew;
      float sum = 0.0f;
      #pragma unroll
      for (int nt = 0; nt < 4; ++nt) {
        float e = expf(p[nt][r] - mnew);
        p[nt][r] = e;
        sum += e;
      }
      rs[r] = sum;
    }
    #pragma unroll
    for (int r = 0; r < 8; ++r)
      #pragma unroll
      for (int off = 1; off < 16; off <<= 1)
        rs[r] += __shfl_xor(rs[r], off);
    #pragma unroll
    for (int r = 0; r < 8; ++r) {
      l[r] = l[r] * corr[r] + rs[r];
      #pragma unroll
      for (int dt = 0; dt < 4; ++dt) oacc[dt][r] *= corr[r];
    }

    // ---- P -> LDS (per-wave private band), same-wave RAW via dscnt ----
    #pragma unroll
    for (int nt = 0; nt < 4; ++nt)
      #pragma unroll
      for (int r = 0; r < 8; ++r)
        Ps[(wq * 16 + (hsel ? 8 + r : r)) * AT_LD + nt * 16 + l15] = (_Float16)p[nt][r];
    asm volatile("s_wait_dscnt 0" ::: "memory");

    // ---- O += P @ V ----
    #pragma unroll
    for (int kd = 0; kd < 2; ++kd) {
      h16 pa = ld_frag_gap(&Ps[(wq * 16 + l15) * AT_LD + kd * 32 + hsel * 16]);
      #pragma unroll
      for (int dt = 0; dt < 4; ++dt) {
        h16 vf = ld_frag_ct(&Vs[(dt * 16 + l15) * AT_LD + kd * 32 + hsel * 16]);
        oacc[dt] = wmma_f16(pa, vf, oacc[dt]);
      }
    }
  }

  #pragma unroll
  for (int dt = 0; dt < 4; ++dt)
    #pragma unroll
    for (int r = 0; r < 8; ++r) {
      int rr = hsel ? 8 + r : r;
      size_t row = (size_t)(qrow0 + wq * 16 + rr);
      O[row * C_D + hc + dt * 16 + l15] = (_Float16)(oacc[dt][r] / l[r]);
    }
}

// ---------------------------------------------------------------------------
// Residual + LayerNorm (ddof=1, eps on std); writes f32 out + f16 mirror.
// ---------------------------------------------------------------------------
__global__ __launch_bounds__(256) void add_ln_kernel(
    const float* __restrict__ X, const float* __restrict__ Y,
    const float* __restrict__ g, const float* __restrict__ be,
    float* __restrict__ Out, _Float16* __restrict__ OutH)
{
  const int row = blockIdx.x;
  const int tid = threadIdx.x;
  float v[4];
  float s = 0.0f, s2 = 0.0f;
  #pragma unroll
  for (int i = 0; i < 4; ++i) {
    int c = tid + i * 256;
    float x = X[(size_t)row * C_D + c] + Y[(size_t)row * C_D + c];
    v[i] = x; s += x; s2 += x * x;
  }
  #pragma unroll
  for (int off = 16; off; off >>= 1) {
    s  += __shfl_xor(s, off);
    s2 += __shfl_xor(s2, off);
  }
  __shared__ float sh1[8], sh2[8];
  int wv = tid >> 5, lane = tid & 31;
  if (lane == 0) { sh1[wv] = s; sh2[wv] = s2; }
  __syncthreads();
  if (tid == 0) {
    float a = 0.0f, c2 = 0.0f;
    #pragma unroll
    for (int i = 0; i < 8; ++i) { a += sh1[i]; c2 += sh2[i]; }
    sh1[0] = a; sh2[0] = c2;
  }
  __syncthreads();
  s = sh1[0]; s2 = sh2[0];
  float mean = s * (1.0f / 1024.0f);
  float var  = fmaxf((s2 - 1024.0f * mean * mean) * (1.0f / 1023.0f), 0.0f);
  float inv  = 1.0f / (sqrtf(var) + 1e-6f);
  #pragma unroll
  for (int i = 0; i < 4; ++i) {
    int c = tid + i * 256;
    float o = g[c] * (v[i] - mean) * inv + be[c];
    Out[(size_t)row * C_D + c]  = o;
    OutH[(size_t)row * C_D + c] = (_Float16)o;
  }
}

// ---------------------------------------------------------------------------
// Host orchestration
// ---------------------------------------------------------------------------
extern "C" void kernel_launch(void* const* d_in, const int* in_sizes, int n_in,
                              void* d_out, int out_size, void* d_ws, size_t ws_size,
                              hipStream_t stream) {
  (void)in_sizes; (void)n_in; (void)out_size; (void)ws_size;

  const float* x_in     = (const float*)d_in[0];
  const float* encoding = (const float*)d_in[1];
  const float* mask_src = (const float*)d_in[2];
  const float* mask_trg = (const float*)d_in[3];
  const float* const* P = (const float* const*)(d_in + 4);
  // P: 0 sa_wq 1 sa_bq 2 sa_wk 3 sa_bk 4 sa_wv 5 sa_bv 6 sa_wo 7 sa_bo 8 sa_g
  //    9 sa_be 10..19 ca_* 20 ff_w1 21 ff_b1 22 ff_w2 23 ff_b2 24 ff_g 25 ff_be

  const size_t MS  = (size_t)C_M * C_D;     // 4M
  const size_t SZ1 = (size_t)C_D * C_D;     // 1M
  float* f32b = (float*)d_ws;
  float* bufA = f32b;                       // x (f32)
  float* bufB = f32b + MS;                  // branch output (f32)
  _Float16* hb = (_Float16*)(f32b + 2 * MS);
  _Float16* xh   = hb;                      // 4M
  _Float16* qh   = hb + 1 * MS;
  _Float16* kh   = hb + 2 * MS;
  _Float16* vh   = hb + 3 * MS;
  _Float16* atth = hb + 4 * MS;
  _Float16* hh   = hb + 5 * MS;             // 16M (M x H)
  _Float16* ench = hb + 9 * MS;             // 24M (6 layers)
  _Float16* wT   = hb + 15 * MS;            // 96M transposed f16 weights
  _Float16* saT[4] = { wT, wT + 6 * SZ1, wT + 12 * SZ1, wT + 18 * SZ1 };
  _Float16* caT[4] = { wT + 24 * SZ1, wT + 30 * SZ1, wT + 36 * SZ1, wT + 42 * SZ1 };
  _Float16* w1T = wT + 48 * SZ1;            // 6 * 4M
  _Float16* w2T = wT + 72 * SZ1;            // 6 * 4M

  // ---- prep: x copy+mirror, enc -> f16, weights -> f16 transposed ----
  init_x_kernel<<<MS / 1024, 256, 0, stream>>>(x_in, bufA, xh);
  cvt_f16_kernel<<<(6 * MS) / 1024, 256, 0, stream>>>(encoding + MS, ench);
  const int wIdx[8] = {0, 2, 4, 6, 10, 12, 14, 16};
  _Float16* wDst[8] = {saT[0], saT[1], saT[2], saT[3], caT[0], caT[1], caT[2], caT[3]};
  for (int i = 0; i < 8; ++i)
    transpose_cvt_kernel<<<dim3(C_D / 32, C_D / 32, 6), 256, 0, stream>>>(
        P[wIdx[i]], wDst[i], C_D, C_D);
  transpose_cvt_kernel<<<dim3(C_H / 32, C_D / 32, 6), 256, 0, stream>>>(P[20], w1T, C_D, C_H);
  transpose_cvt_kernel<<<dim3(C_D / 32, C_H / 32, 6), 256, 0, stream>>>(P[22], w2T, C_H, C_D);

  dim3 gD(C_D / GM_BN, C_M / GM_BM);        // (8, 32)
  dim3 gH(C_H / GM_BN, C_M / GM_BM);        // (32, 32)
  dim3 gA(C_T / 64, C_B * C_NH);            // (16, 64)

  for (int lyr = 0; lyr < 6; ++lyr) {
    const size_t bOff = (size_t)lyr * C_D;
    const size_t wO   = (size_t)lyr * SZ1;
    const size_t wHO  = (size_t)lyr * 4 * SZ1;
    _Float16* encl = ench + (size_t)lyr * MS;

    // ---- self attention ----
    gemm_kernel<0,1><<<gD, 256, 0, stream>>>(xh, saT[0] + wO, P[1] + bOff, nullptr, qh, C_M, C_D, C_D);
    gemm_kernel<0,1><<<gD, 256, 0, stream>>>(xh, saT[1] + wO, P[3] + bOff, nullptr, kh, C_M, C_D, C_D);
    gemm_kernel<0,1><<<gD, 256, 0, stream>>>(xh, saT[2] + wO, P[5] + bOff, nullptr, vh, C_M, C_D, C_D);
    attn_kernel<<<gA, 128, 0, stream>>>(qh, kh, vh, mask_trg, atth, 1);
    gemm_kernel<0,0><<<gD, 256, 0, stream>>>(atth, saT[3] + wO, P[7] + bOff, bufB, nullptr, C_M, C_D, C_D);
    add_ln_kernel<<<C_M, 256, 0, stream>>>(bufA, bufB, P[8] + bOff, P[9] + bOff, bufA, xh);

    // ---- cross attention ----
    gemm_kernel<0,1><<<gD, 256, 0, stream>>>(xh,   caT[0] + wO, P[11] + bOff, nullptr, qh, C_M, C_D, C_D);
    gemm_kernel<0,1><<<gD, 256, 0, stream>>>(encl, caT[1] + wO, P[13] + bOff, nullptr, kh, C_M, C_D, C_D);
    gemm_kernel<0,1><<<gD, 256, 0, stream>>>(encl, caT[2] + wO, P[15] + bOff, nullptr, vh, C_M, C_D, C_D);
    attn_kernel<<<gA, 128, 0, stream>>>(qh, kh, vh, mask_src, atth, 0);
    gemm_kernel<0,0><<<gD, 256, 0, stream>>>(atth, caT[3] + wO, P[17] + bOff, bufB, nullptr, C_M, C_D, C_D);
    add_ln_kernel<<<C_M, 256, 0, stream>>>(bufA, bufB, P[18] + bOff, P[19] + bOff, bufA, xh);

    // ---- feed forward ----
    gemm_kernel<1,1><<<gH, 256, 0, stream>>>(xh, w1T + wHO, P[21] + (size_t)lyr * C_H, nullptr, hh, C_M, C_D, C_H);
    gemm_kernel<0,0><<<gD, 256, 0, stream>>>(hh, w2T + wHO, P[23] + bOff, bufB, nullptr, C_M, C_H, C_D);
    float* outp = (lyr == 5) ? (float*)d_out : bufA;
    add_ln_kernel<<<C_M, 256, 0, stream>>>(bufA, bufB, P[24] + bOff, P[25] + bOff, outp, xh);
  }
}
